// LSTMAutoencoder_3908420239502
// MI455X (gfx1250) — compile-verified, matrix operand order checked by reading, not run on hardware
//
#include <hip/hip_runtime.h>

// ---------------------------------------------------------------------------
// LSTM autoencoder for MI455X (gfx1250), wave32 + v_wmma_f32_16x16x32_bf16.
// B=512, T=256, D=H=128, gates=4H=512.
//
// Each of 32 workgroups (1024 threads = 32 waves) owns 16 batch rows and runs
// the whole encoder+decoder with only workgroup barriers (batch rows are
// independent end-to-end). Per cell step each wave computes one 16x16 gate
// tile: K=256 -> 8 bf16 WMMAs (f32 accumulate). Weights are pre-packed into
// WMMA B-fragment lane order; the compiler keeps the two active cells'
// B fragments VGPR-resident across the time loop (A is loaded interleaved
// from LDS to keep register pressure below the spill threshold).
// Activations are hardware transcendentals, branch-free on the serial path.
// ---------------------------------------------------------------------------

typedef __attribute__((ext_vector_type(16))) __bf16 v16bf;
typedef __attribute__((ext_vector_type(8)))  float  v8f;

#define B_  512
#define T_  256
#define D_  128
#define H_  128
#define NG  512   // 4*H gate width

#define LOG2E 1.4426950408889634f

__device__ inline float fast_sigmoid(float x) {
  // 1/(1+2^(-x*log2e)) : v_exp_f32 + v_rcp_f32, branch-free
  return __builtin_amdgcn_rcpf(1.f + __builtin_amdgcn_exp2f(-LOG2E * x));
}

__device__ inline float fast_tanh(float x) {
#if __has_builtin(__builtin_amdgcn_tanhf)
  return __builtin_amdgcn_tanhf(x);
#elif __has_builtin(__builtin_amdgcn_tanh_f32)
  return __builtin_amdgcn_tanh_f32(x);
#else
  // tanh(x) = 2*sigmoid(2x) - 1, branch-free trans-op fallback
  return 2.f * __builtin_amdgcn_rcpf(1.f + __builtin_amdgcn_exp2f(-2.f * LOG2E * x)) - 1.f;
#endif
}

// 16-bit A/B fragment element position (ISA 7.12.2, 16-bit 16x32 A layout;
// B 32x16 mirrors it with lane<->N). Offset within [kb][lane][16] blob.
__device__ __host__ inline int frag_pos(int mn /*0..15*/, int k) {
  int kb = k >> 5;
  int kl = k & 31;
  int half, v, s;
  if (kl < 16) { half = kl >> 3; v = (kl & 7) >> 1; s = kl & 1; }
  else { int t = kl - 16; half = t >> 3; v = 4 + ((t & 7) >> 1); s = t & 1; }
  int lane = (half << 4) | mn;
  return kb * 512 + lane * 16 + v * 2 + s;
}

// ---- prep: pack [Wih^T ; Whh^T] (K=256 x N=512) into bf16 B fragments ----
__global__ void prep_cellW(const float* __restrict__ Wih,
                           const float* __restrict__ Whh,
                           const float* __restrict__ bih,
                           const float* __restrict__ bhh,
                           __bf16* __restrict__ Wout,   // 32 ntiles * 8 kb * 512
                           float*  __restrict__ bout) { // 512 combined bias
  int idx = blockIdx.x * blockDim.x + threadIdx.x;      // 0..131071
  if (idx < 131072) {
    int n_tile = idx >> 12;
    int rem    = idx & 4095;
    int kb     = rem >> 9;
    int r2     = rem & 511;
    int lane   = r2 >> 4;
    int e      = r2 & 15;
    int v = e >> 1, s = e & 1, half = lane >> 4;
    int kl = (v < 4) ? (half * 8 + v * 2 + s) : (16 + half * 8 + (v - 4) * 2 + s);
    int k = kb * 32 + kl;
    int n = n_tile * 16 + (lane & 15);
    float w = (k < 128) ? Wih[n * 128 + k] : Whh[n * 128 + (k - 128)];
    Wout[idx] = (__bf16)w;
  }
  if (idx < 512) bout[idx] = bih[idx] + bhh[idx];
}

// ---- prep: pack out_W^T (K=128 x N=128) into bf16 B fragments ----
__global__ void prep_outW(const float* __restrict__ Wsrc,  // (D,H) row-major
                          const float* __restrict__ bsrc,  // (D,)
                          __bf16* __restrict__ Wout,       // 8 ntiles * 4 kb * 512
                          float*  __restrict__ bout) {
  int idx = blockIdx.x * blockDim.x + threadIdx.x;         // 0..16383
  if (idx < 16384) {
    int n_tile = idx >> 11;
    int rem    = idx & 2047;
    int kb     = rem >> 9;
    int r2     = rem & 511;
    int lane   = r2 >> 4;
    int e      = r2 & 15;
    int v = e >> 1, s = e & 1, half = lane >> 4;
    int kl = (v < 4) ? (half * 8 + v * 2 + s) : (16 + half * 8 + (v - 4) * 2 + s);
    int k = kb * 32 + kl;
    int n = n_tile * 16 + (lane & 15);
    Wout[idx] = (__bf16)Wsrc[n * H_ + k];
  }
  if (idx < 128) bout[idx] = bsrc[idx];
}

// One 16x16 gate tile: C = A(16x256) * B(256x512 col slice) + bias -> gates LDS.
// A loaded interleaved (keeps pressure low so B stays VGPR-resident; cross-
// wave occupancy of 8 waves/SIMD hides the per-wave ds latency).
__device__ inline void gemm_tile(const __bf16* __restrict__ Alds,
                                 const __bf16* __restrict__ Bglob,
                                 const float* __restrict__ bias,
                                 float* __restrict__ gates,
                                 int wave, int lane) {
  v8f acc = {};
  #pragma unroll
  for (int kb = 0; kb < 8; ++kb) {
    v16bf a = *(const v16bf*)(Alds + kb * 512 + lane * 16);
    v16bf b = *(const v16bf*)(Bglob + (wave * 8 + kb) * 512 + lane * 16);
    acc = __builtin_amdgcn_wmma_f32_16x16x32_bf16(false, a, false, b,
                                                  (short)0, acc, false, false);
  }
  int n = wave * 16 + (lane & 15);
  float bv = bias[n];
  int mbase = (lane >> 4) * 8;           // C layout: vgpr r, lanes>=16 -> M=r+8
  #pragma unroll
  for (int r = 0; r < 8; ++r) gates[(mbase + r) * NG + n] = acc[r] + bv;
}

__device__ inline void cell_update(const float* __restrict__ gates,
                                   float* c, int m, int j0, float* h) {
  #pragma unroll
  for (int e = 0; e < 2; ++e) {
    int j = j0 + e;
    float gi = gates[m * NG + j];
    float gf = gates[m * NG + 128 + j];
    float gg = gates[m * NG + 256 + j];
    float go = gates[m * NG + 384 + j];
    float cn = fast_sigmoid(gf) * c[e] + fast_sigmoid(gi) * fast_tanh(gg);
    c[e] = cn;
    h[e] = fast_sigmoid(go) * fast_tanh(cn);
  }
}

__global__ __launch_bounds__(1024, 1)
void lstm_ae_kernel(const float* __restrict__ x,
                    const __bf16* __restrict__ We0, const __bf16* __restrict__ We1,
                    const __bf16* __restrict__ Wd0, const __bf16* __restrict__ Wd1,
                    const __bf16* __restrict__ Wo,
                    const float* __restrict__ be0, const float* __restrict__ be1,
                    const float* __restrict__ bd0, const float* __restrict__ bd1,
                    const float* __restrict__ bo,
                    float* __restrict__ out) {
  // A buffers hold [x_t | h_{t-1}] (16x256 bf16) in fragment layout.
  __shared__ __align__(32) __bf16 A0[4096];
  __shared__ __align__(32) __bf16 A1[4096];
  __shared__ float gates[16 * NG];       // 32 KB

  const int tid  = threadIdx.x;
  const int lane = tid & 31;
  const int wave = tid >> 5;             // == n_tile, 0..31
  const int b0   = blockIdx.x * 16;

  // cell-update ownership: thread -> (row m_u, hidden units j_u, j_u+1)
  const int m_u = (tid * 2) >> 7;
  const int j_u = (tid * 2) & 127;

  float c_e0[2] = {0.f, 0.f}, c_e1[2] = {0.f, 0.f};
  float c_d0[2] = {0.f, 0.f}, c_d1[2] = {0.f, 0.f};
  float h_keep[2] = {0.f, 0.f};
  float h[2];

  for (int i = tid; i < 4096; i += 1024) { A0[i] = (__bf16)0.f; A1[i] = (__bf16)0.f; }
  __syncthreads();

  // ================= encoder (2 layers, fused per step) =================
  for (int t = 0; t < T_; ++t) {
    {   // stage x_t into A0 x-part (bf16, fragment layout)
      int m = m_u, d = j_u;
      const float2 xv = *(const float2*)(x + (((size_t)(b0 + m)) * T_ + t) * D_ + d);
      A0[frag_pos(m, d)]     = (__bf16)xv.x;
      A0[frag_pos(m, d + 1)] = (__bf16)xv.y;
    }
    __syncthreads();
    gemm_tile(A0, We0, be0, gates, wave, lane);
    __syncthreads();
    cell_update(gates, c_e0, m_u, j_u, h);
    A0[frag_pos(m_u, 128 + j_u)]     = (__bf16)h[0];   // enc0 feedback
    A0[frag_pos(m_u, 128 + j_u + 1)] = (__bf16)h[1];
    A1[frag_pos(m_u, j_u)]           = (__bf16)h[0];   // enc1 input
    A1[frag_pos(m_u, j_u + 1)]       = (__bf16)h[1];
    __syncthreads();
    gemm_tile(A1, We1, be1, gates, wave, lane);
    __syncthreads();
    cell_update(gates, c_e1, m_u, j_u, h);
    A1[frag_pos(m_u, 128 + j_u)]     = (__bf16)h[0];   // enc1 feedback
    A1[frag_pos(m_u, 128 + j_u + 1)] = (__bf16)h[1];
    if (t == T_ - 1) { h_keep[0] = h[0]; h_keep[1] = h[1]; }
  }
  __syncthreads();

  // ======= transition: decoder state zero, x_in = encoded[:, -1, :] =======
  for (int i = tid; i < 4096; i += 1024) { A0[i] = (__bf16)0.f; A1[i] = (__bf16)0.f; }
  __syncthreads();
  A0[frag_pos(m_u, j_u)]     = (__bf16)h_keep[0];
  A0[frag_pos(m_u, j_u + 1)] = (__bf16)h_keep[1];
  __syncthreads();

  // ================= decoder (2 cells + out proj per step) =================
  for (int t = 0; t < T_; ++t) {
    gemm_tile(A0, Wd0, bd0, gates, wave, lane);
    __syncthreads();
    cell_update(gates, c_d0, m_u, j_u, h);
    A0[frag_pos(m_u, 128 + j_u)]     = (__bf16)h[0];   // dec0 feedback
    A0[frag_pos(m_u, 128 + j_u + 1)] = (__bf16)h[1];
    A1[frag_pos(m_u, j_u)]           = (__bf16)h[0];   // dec1 input
    A1[frag_pos(m_u, j_u + 1)]       = (__bf16)h[1];
    __syncthreads();
    gemm_tile(A1, Wd1, bd1, gates, wave, lane);
    __syncthreads();
    cell_update(gates, c_d1, m_u, j_u, h);
    A1[frag_pos(m_u, 128 + j_u)]     = (__bf16)h[0];   // dec1 feedback
    A1[frag_pos(m_u, 128 + j_u + 1)] = (__bf16)h[1];
    A0[frag_pos(m_u, j_u)]           = (__bf16)h[0];   // x_in(next) = h1
    A0[frag_pos(m_u, j_u + 1)]       = (__bf16)h[1];
    __syncthreads();
    // out = h1 @ out_W^T + out_b ; reuse A0 x-part (== h1) as A operand.
    if (wave < 8) {
      v8f acc = {};
      #pragma unroll
      for (int kb = 0; kb < 4; ++kb) {
        v16bf a = *(const v16bf*)(A0 + kb * 512 + lane * 16);
        v16bf b = *(const v16bf*)(Wo + (wave * 4 + kb) * 512 + lane * 16);
        acc = __builtin_amdgcn_wmma_f32_16x16x32_bf16(false, a, false, b,
                                                      (short)0, acc, false, false);
      }
      int n = wave * 16 + (lane & 15);
      float bv = bo[n];
      int mbase = (lane >> 4) * 8;
      #pragma unroll
      for (int r = 0; r < 8; ++r) {
        int m = mbase + r;
        out[(((size_t)(b0 + m)) * T_ + t) * D_ + n] = acc[r] + bv;
      }
    }
    // other waves proceed into next GEMM (reads only; first barrier joins all)
  }
}

extern "C" void kernel_launch(void* const* d_in, const int* in_sizes, int n_in,
                              void* d_out, int out_size, void* d_ws, size_t ws_size,
                              hipStream_t stream) {
  const float* x = (const float*)d_in[0];
  // dict order: enc0(Wih,Whh,bih,bhh), enc1(...), dec0(...), dec1(...), out_W, out_b
  const float* Wih[4] = {(const float*)d_in[1],  (const float*)d_in[5],
                         (const float*)d_in[9],  (const float*)d_in[13]};
  const float* Whh[4] = {(const float*)d_in[2],  (const float*)d_in[6],
                         (const float*)d_in[10], (const float*)d_in[14]};
  const float* bih[4] = {(const float*)d_in[3],  (const float*)d_in[7],
                         (const float*)d_in[11], (const float*)d_in[15]};
  const float* bhh[4] = {(const float*)d_in[4],  (const float*)d_in[8],
                         (const float*)d_in[12], (const float*)d_in[16]};
  const float* outW = (const float*)d_in[17];
  const float* outb = (const float*)d_in[18];

  // workspace layout (bytes): 4 packed cell weight blobs (256 KB each),
  // packed out_W (32 KB), 4 combined biases (2 KB each), out bias.
  char* ws = (char*)d_ws;
  __bf16* Wc[4];
  float*  bc[4];
  for (int c = 0; c < 4; ++c) {
    Wc[c] = (__bf16*)(ws + (size_t)c * 262144);
    bc[c] = (float*)(ws + 1081344 + (size_t)c * 2048);
  }
  __bf16* Wo = (__bf16*)(ws + 1048576);
  float*  bo = (float*)(ws + 1089536);

  for (int c = 0; c < 4; ++c)
    prep_cellW<<<512, 256, 0, stream>>>(Wih[c], Whh[c], bih[c], bhh[c], Wc[c], bc[c]);
  prep_outW<<<64, 256, 0, stream>>>(outW, outb, Wo, bo);

  lstm_ae_kernel<<<32, 1024, 0, stream>>>(x, Wc[0], Wc[1], Wc[2], Wc[3], Wo,
                                          bc[0], bc[1], bc[2], bc[3], bo,
                                          (float*)d_out);
}
